// NIUcausal_multi_head_self_attention_41412074668318
// MI455X (gfx1250) — matrix operand order repaired and checked by reading, not compiled, not run on hardware
//
#include <hip/hip_runtime.h>

typedef __attribute__((ext_vector_type(16))) _Float16 v16h;
typedef __attribute__((ext_vector_type(8)))  float    v8f;

union F16Frag { v16h v; unsigned u[8]; _Float16 h[16]; };
union H4      { _Float16 h[4]; uint2 u; };

#define WMMA_F16(A, B, C) \
    __builtin_amdgcn_wmma_f32_16x16x32_f16(false, (A), false, (B), (short)0, (C), false, false)

// ---- TDM availability hedging (per-pass; host pass simply compiles fallback)
#if defined(__has_builtin)
#if __has_builtin(__builtin_amdgcn_tensor_load_to_lds)
#define HAVE_TDM 1
#endif
#endif
#ifndef HAVE_TDM
#define HAVE_TDM 0
#endif
#if __has_include(<hip/amd_detail/amd_gfx1250_TDM.h>)
#define TDM_ARITY6 1
#else
#define TDM_ARITY6 0
#endif

// Issue a TDM load of a 64x64 f16 tile (row stride 1024 elements) into LDS,
// adding 2 dwords of pad every 32 dwords (-> LDS rows of 68 halfs).
__device__ __forceinline__ void tdm_load_tile64x64(const _Float16* gsrc, unsigned lds_byte_off)
{
#if HAVE_TDM
    typedef __attribute__((ext_vector_type(4))) unsigned int u32x4;
    typedef __attribute__((ext_vector_type(8))) int          i32x8;
    typedef __attribute__((ext_vector_type(4))) int          i32x4;
    unsigned long long ga = (unsigned long long)(const void*)gsrc;
    // D# group0: count=1 | lds_addr | global_addr[56:0] | type=2
    u32x4 g0 = (u32x4){ 1u,
                        lds_byte_off,
                        (unsigned)ga,
                        (unsigned)((ga >> 32) & 0x1FFFFFFull) | (2u << 30) };
    // D# group1: data_size=2B, pad_enable, pad_interval=32dw, pad_amount=2dw,
    //            tensor_dim0=64, tensor_dim1=64, tile_dim0=64, tile_dim1=64,
    //            tensor_dim0_stride=1024
    i32x8 g1 = (i32x8){ (int)((1u << 16) | (1u << 20) | (4u << 22) | (1u << 25)),
                        (int)(64u << 16),   // tensor_dim0 lo16 in [63:48]
                        (int)(64u << 16),   // tensor_dim0 hi=0, tensor_dim1 lo16
                        (int)(64u << 16),   // tensor_dim1 hi=0, tile_dim0=64
                        64,                 // tile_dim1=64, tile_dim2=0
                        1024,               // tensor_dim0_stride lo32
                        0, 0 };
    i32x4 z4 = (i32x4){0, 0, 0, 0};
#if TDM_ARITY6
    i32x8 z8 = (i32x8){0, 0, 0, 0, 0, 0, 0, 0};
    __builtin_amdgcn_tensor_load_to_lds(g0, g1, z4, z4, z8, 0);
#else
    __builtin_amdgcn_tensor_load_to_lds(g0, g1, z4, z4, 0);
#endif
#endif
}

__device__ __forceinline__ void tdm_wait()
{
#if HAVE_TDM
    __builtin_amdgcn_s_wait_tensorcnt(0);
#endif
}

// ---------------------------------------------------------------------------
// Tiled GEMM: C[M,N] = A[M,K] * W[K,N], f16 WMMA, f32 accumulation.
// Block = 256 threads (8 waves); tile 128x64, BK=32; wave owns 32x32 (4 WMMA).
// Software-pipelined: next global tile loads into VGPRs during compute.
// ---------------------------------------------------------------------------
template<bool A_HALF, bool OUT_HALF>
__global__ __launch_bounds__(256)
void gemm_wmma_kernel(const void* __restrict__ Aptr, const float* __restrict__ W,
                      void* __restrict__ Cptr, int M, int N, int K)
{
    __shared__ __align__(16) _Float16 As[128][40];  // [m][k], padded
    __shared__ __align__(16) _Float16 Bt[64][40];   // transposed: [n][k], padded

    const int tid  = threadIdx.x;
    const int wave = tid >> 5;
    const int lane = tid & 31;
    const int half = lane >> 4;
    const int l16  = lane & 15;

    const int m0 = blockIdx.y * 128;
    const int n0 = blockIdx.x * 64;
    const int wr = wave & 3;     // 4 row groups of 32
    const int wc = wave >> 2;    // 2 col groups of 32

    v8f acc[2][2];
    #pragma unroll
    for (int i = 0; i < 2; ++i)
        #pragma unroll
        for (int j = 0; j < 2; ++j) { v8f z = {}; acc[i][j] = z; }

    float4 aF[4]; uint2 aH[4]; float4 wF[2];

    auto loadTiles = [&](int k0) {
        #pragma unroll
        for (int e = 0; e < 4; ++e) {           // A: 128x32 (1024 x vec4)
            int v = tid + e * 256; int r = v >> 3; int c4 = (v & 7) << 2;
            if (A_HALF) aH[e] = *(const uint2*)((const _Float16*)Aptr + (size_t)(m0 + r) * K + k0 + c4);
            else        aF[e] = *(const float4*)((const float*)Aptr + (size_t)(m0 + r) * K + k0 + c4);
        }
        #pragma unroll
        for (int e = 0; e < 2; ++e) {           // W: 32x64 (512 x vec4)
            int v = tid + e * 256; int r = v >> 4; int c4 = (v & 15) << 2;
            wF[e] = *(const float4*)(W + (size_t)(k0 + r) * N + n0 + c4);
        }
    };
    auto storeTiles = [&]() {
        #pragma unroll
        for (int e = 0; e < 4; ++e) {
            int v = tid + e * 256; int r = v >> 3; int c4 = (v & 7) << 2;
            if (A_HALF) *(uint2*)&As[r][c4] = aH[e];
            else {
                H4 t;
                t.h[0] = (_Float16)aF[e].x; t.h[1] = (_Float16)aF[e].y;
                t.h[2] = (_Float16)aF[e].z; t.h[3] = (_Float16)aF[e].w;
                *(uint2*)&As[r][c4] = t.u;
            }
        }
        #pragma unroll
        for (int e = 0; e < 2; ++e) {           // scatter into transposed Bt
            int v = tid + e * 256; int r = v >> 4; int c4 = (v & 15) << 2;
            Bt[c4 + 0][r] = (_Float16)wF[e].x;
            Bt[c4 + 1][r] = (_Float16)wF[e].y;
            Bt[c4 + 2][r] = (_Float16)wF[e].z;
            Bt[c4 + 3][r] = (_Float16)wF[e].w;
        }
    };

    loadTiles(0);
    for (int k0 = 0; k0 < K; k0 += 32) {
        storeTiles();
        __syncthreads();
        if (k0 + 32 < K) loadTiles(k0 + 32);

        // A fragments (ISA A-layout swizzle), contiguous b32 LDS reads
        F16Frag a0, a1;
        #pragma unroll
        for (int i = 0; i < 8; ++i) {
            int kk = ((i >> 2) << 4) + half * 8 + ((i & 3) << 1);
            a0.u[i] = *(const unsigned*)&As[wr * 32 + l16][kk];
            a1.u[i] = *(const unsigned*)&As[wr * 32 + 16 + l16][kk];
        }
        // B fragments from transposed tile (ISA B-layout: k = (i>>2)*16 + half*8 + 2*(i&3))
        F16Frag b0, b1;
        #pragma unroll
        for (int i = 0; i < 8; ++i) {
            int kb = ((i >> 2) << 4) + half * 8 + ((i & 3) << 1);
            b0.u[i] = *(const unsigned*)&Bt[wc * 32 + l16][kb];
            b1.u[i] = *(const unsigned*)&Bt[wc * 32 + 16 + l16][kb];
        }
        acc[0][0] = WMMA_F16(a0.v, b0.v, acc[0][0]);
        acc[0][1] = WMMA_F16(a0.v, b1.v, acc[0][1]);
        acc[1][0] = WMMA_F16(a1.v, b0.v, acc[1][0]);
        acc[1][1] = WMMA_F16(a1.v, b1.v, acc[1][1]);
        __syncthreads();
    }

    // C/D layout: VGPR r holds (M = r + 8*half, N = l16)
    #pragma unroll
    for (int sm = 0; sm < 2; ++sm)
        #pragma unroll
        for (int r = 0; r < 8; ++r) {
            int row = m0 + wr * 32 + sm * 16 + r + half * 8;
            #pragma unroll
            for (int sn = 0; sn < 2; ++sn) {
                int col = n0 + wc * 32 + sn * 16 + l16;
                if (OUT_HALF)
                    ((_Float16*)Cptr)[(size_t)row * N + col] = (_Float16)acc[sm][sn][r];
                else
                    ((float*)Cptr)[(size_t)row * N + col] = acc[sm][sn][r];
            }
        }
}

// ---------------------------------------------------------------------------
// RoPE in place on Q and K (f16 workspace). One thread per (row, pair).
// ---------------------------------------------------------------------------
__global__ __launch_bounds__(256)
void rope_kernel(_Float16* __restrict__ Q, _Float16* __restrict__ Kk,
                 const int* __restrict__ pos)
{
    const int S = 2048, D = 1024;
    int t = blockIdx.x * 256 + threadIdx.x;      // total = 8192 * 512
    int row  = t >> 9;
    int j    = t & 511;
    int head = j >> 5;
    int jl   = j & 31;
    int s = row & (S - 1);
    float p = (float)pos[s];
    float inv = __expf(-(float)jl * 0.28782313662425572f);   // ln(1e4)/32
    float ang = p * inv;
    float sn, cs;
    __sincosf(ang, &sn, &cs);
    size_t base = (size_t)row * D + head * 64 + 2 * jl;

    float q1 = (float)Q[base], q2 = (float)Q[base + 1];
    Q[base]     = (_Float16)(q1 * cs - q2 * sn);
    Q[base + 1] = (_Float16)(q1 * sn + q2 * cs);

    float k1 = (float)Kk[base], k2 = (float)Kk[base + 1];
    Kk[base]     = (_Float16)(k1 * cs - k2 * sn);
    Kk[base + 1] = (_Float16)(k1 * sn + k2 * cs);
}

// Manual fallback staging of a 64x64 f16 tile (row stride 1024) into [64][68] LDS.
__device__ __forceinline__ void stage_tile_rowmajor(_Float16 (*dst)[68],
                                                    const _Float16* src, int tid)
{
    #pragma unroll
    for (int e = 0; e < 8; ++e) {
        int v = tid + e * 128; int r = v >> 4; int c4 = (v & 15) << 2;
        *(uint2*)&dst[r][c4] = *(const uint2*)(src + (size_t)r * 1024 + c4);
    }
}

// ---------------------------------------------------------------------------
// Flash attention: grid (S/64, B*H), 128 threads (4 waves), wave owns 16 q-rows.
// Q/K tiles via TDM (with LDS padding), V staged transposed, P via LDS.
// ---------------------------------------------------------------------------
__global__ __launch_bounds__(128)
void flash_attn_kernel(const _Float16* __restrict__ Q, const _Float16* __restrict__ Kk,
                       const _Float16* __restrict__ V, _Float16* __restrict__ O)
{
    const int S = 2048, D = 1024;
    __shared__ __align__(16) _Float16 Qs[64][68];
    __shared__ __align__(16) _Float16 Ks[64][68];
    __shared__ __align__(16) _Float16 Vt[64][68];   // transposed: [hd][key]
    __shared__ __align__(16) _Float16 Ps[4][16][68];

    const int qt = blockIdx.x;
    const int bh = blockIdx.y;
    const int b  = bh >> 4;
    const int h  = bh & 15;

    const int tid  = threadIdx.x;
    const int wave = tid >> 5;
    const int lane = tid & 31;
    const int half = lane >> 4;
    const int l16  = lane & 15;

    const size_t rowbase = (size_t)(b * S + qt * 64);
    const _Float16* Qsrc = Q + rowbase * D + h * 64;

    // Stage Q tile
#if HAVE_TDM
    if (wave == 0) {
        tdm_load_tile64x64(Qsrc, (unsigned)(unsigned long long)(void*)&Qs[0][0]);
        tdm_wait();
    }
#else
    stage_tile_rowmajor(Qs, Qsrc, tid);
#endif
    __syncthreads();

    // Hoisted Q A-fragments for this wave's 16 rows (d chunks 0..31 / 32..63)
    F16Frag aq0, aq1;
    #pragma unroll
    for (int i = 0; i < 8; ++i) {
        int kk = ((i >> 2) << 4) + half * 8 + ((i & 3) << 1);
        aq0.u[i] = *(const unsigned*)&Qs[wave * 16 + l16][kk];
        aq1.u[i] = *(const unsigned*)&Qs[wave * 16 + l16][32 + kk];
    }

    v8f acc[4];
    #pragma unroll
    for (int n = 0; n < 4; ++n) { v8f z = {}; acc[n] = z; }
    float m_r[8], l_r[8];
    #pragma unroll
    for (int r = 0; r < 8; ++r) { m_r[r] = -1e30f; l_r[r] = 0.0f; }

    const float scale = 0.125f;  // 1/sqrt(64)

    for (int kt = 0; kt <= qt; ++kt) {
        const _Float16* Ksrc = Kk + (size_t)(b * S + kt * 64) * D + h * 64;
        const _Float16* Vsrc = V  + (size_t)(b * S + kt * 64) * D + h * 64;
#if HAVE_TDM
        if (wave == 0)
            tdm_load_tile64x64(Ksrc, (unsigned)(unsigned long long)(void*)&Ks[0][0]);
#else
        stage_tile_rowmajor(Ks, Ksrc, tid);
#endif
        // All threads stage V transposed while TDM streams K
        #pragma unroll
        for (int e = 0; e < 8; ++e) {
            int v = tid + e * 128; int r = v >> 4; int c4 = (v & 15) << 2;
            H4 t; t.u = *(const uint2*)(Vsrc + (size_t)r * 1024 + c4);
            Vt[c4 + 0][r] = t.h[0]; Vt[c4 + 1][r] = t.h[1];
            Vt[c4 + 2][r] = t.h[2]; Vt[c4 + 3][r] = t.h[3];
        }
#if HAVE_TDM
        if (wave == 0) tdm_wait();
#endif
        __syncthreads();

        // scores = Q * K^T  (B operand: N = key col, K-dim = d; Ks rows contiguous)
        v8f sc[4];
        #pragma unroll
        for (int n = 0; n < 4; ++n) {
            F16Frag bk0, bk1;
            #pragma unroll
            for (int i = 0; i < 8; ++i) {
                int dd = ((i >> 2) << 4) + half * 8 + ((i & 3) << 1);
                bk0.u[i] = *(const unsigned*)&Ks[n * 16 + l16][dd];
                bk1.u[i] = *(const unsigned*)&Ks[n * 16 + l16][32 + dd];
            }
            v8f z = {};
            z = WMMA_F16(aq0.v, bk0.v, z);
            z = WMMA_F16(aq1.v, bk1.v, z);
            sc[n] = z;
        }

        const bool diag = (kt == qt);
        #pragma unroll
        for (int r = 0; r < 8; ++r) {
            int qrow = wave * 16 + r + half * 8;
            float vals[4];
            float rowmax = -1e30f;
            #pragma unroll
            for (int n = 0; n < 4; ++n) {
                float v = sc[n][r] * scale;
                if (diag) { int kcol = n * 16 + l16; if (kcol > qrow) v = -1e30f; }
                vals[n] = v;
                rowmax = fmaxf(rowmax, v);
            }
            #pragma unroll
            for (int off = 1; off < 16; off <<= 1)
                rowmax = fmaxf(rowmax, __shfl_xor(rowmax, off, 32));
            float newm  = fmaxf(m_r[r], rowmax);
            float alpha = __expf(m_r[r] - newm);
            float rsum = 0.0f;
            #pragma unroll
            for (int n = 0; n < 4; ++n) {
                float pv = __expf(vals[n] - newm);
                vals[n] = pv;
                rsum += pv;
            }
            #pragma unroll
            for (int off = 1; off < 16; off <<= 1)
                rsum += __shfl_xor(rsum, off, 32);
            l_r[r] = l_r[r] * alpha + rsum;
            m_r[r] = newm;
            #pragma unroll
            for (int n = 0; n < 4; ++n) {
                acc[n][r] = acc[n][r] * alpha;
                Ps[wave][r + half * 8][n * 16 + l16] = (_Float16)vals[n];
            }
        }

        // Reload P in A-layout (per-wave LDS round trip = lane transpose)
        F16Frag ap0, ap1;
        #pragma unroll
        for (int i = 0; i < 8; ++i) {
            int kk = ((i >> 2) << 4) + half * 8 + ((i & 3) << 1);
            ap0.u[i] = *(const unsigned*)&Ps[wave][l16][kk];
            ap1.u[i] = *(const unsigned*)&Ps[wave][l16][32 + kk];
        }
        // O += P * V   (B operand from transposed Vt: contiguous b32 reads)
        #pragma unroll
        for (int n = 0; n < 4; ++n) {
            F16Frag bv0, bv1;
            #pragma unroll
            for (int i = 0; i < 8; ++i) {
                int kb = ((i >> 2) << 4) + half * 8 + ((i & 3) << 1);
                bv0.u[i] = *(const unsigned*)&Vt[n * 16 + l16][kb];
                bv1.u[i] = *(const unsigned*)&Vt[n * 16 + l16][32 + kb];
            }
            acc[n] = WMMA_F16(ap0.v, bv0.v, acc[n]);
            acc[n] = WMMA_F16(ap1.v, bv1.v, acc[n]);
        }
        __syncthreads();
    }

    // Normalize and write O (f16 workspace)
    #pragma unroll
    for (int r = 0; r < 8; ++r) {
        float invl = 1.0f / l_r[r];
        size_t row = rowbase + wave * 16 + r + half * 8;
        #pragma unroll
        for (int n = 0; n < 4; ++n)
            O[row * D + h * 64 + n * 16 + l16] = (_Float16)(acc[n][r] * invl);
    }
}

// ---------------------------------------------------------------------------
extern "C" void kernel_launch(void* const* d_in, const int* in_sizes, int n_in,
                              void* d_out, int out_size, void* d_ws, size_t ws_size,
                              hipStream_t stream) {
    const int B = 4, S = 2048, D = 1024;
    const int M = B * S;                 // 8192
    const size_t NH = (size_t)M * D;     // elems per f16 buffer (16 MB each)

    const float* x  = (const float*)d_in[0];
    const float* Wq = (const float*)d_in[1];
    const float* Wk = (const float*)d_in[2];
    const float* Wv = (const float*)d_in[3];
    const float* Wo = (const float*)d_in[4];
    const int*   tp = (const int*)d_in[5];

    _Float16* Qh = (_Float16*)d_ws;
    _Float16* Kh = Qh + NH;
    _Float16* Vh = Kh + NH;
    _Float16* Oh = Vh + NH;

    dim3 gemm_grid(D / 64, M / 128);     // (16, 64)

    gemm_wmma_kernel<false, true><<<gemm_grid, 256, 0, stream>>>((const void*)x, Wq, (void*)Qh, M, D, D);
    gemm_wmma_kernel<false, true><<<gemm_grid, 256, 0, stream>>>((const void*)x, Wk, (void*)Kh, M, D, D);
    gemm_wmma_kernel<false, true><<<gemm_grid, 256, 0, stream>>>((const void*)x, Wv, (void*)Vh, M, D, D);

    rope_kernel<<<(M * (D / 2)) / 256, 256, 0, stream>>>(Qh, Kh, tp);

    flash_attn_kernel<<<dim3(S / 64, B * 16), 128, 0, stream>>>(Qh, Kh, Vh, Oh);

    gemm_wmma_kernel<true, false><<<gemm_grid, 256, 0, stream>>>((const void*)Oh, Wo, d_out, M, D, D);
}
